// QBiGRU_65841848648015
// MI455X (gfx1250) — matrix operand order repaired and checked by reading, not compiled, not run on hardware
//
#include <hip/hip_runtime.h>
#include <hip/hip_bf16.h>

#define QDIM     16
#define INPUT_SZ 128
#define HIDDEN   128
#define SEQ      1024
#define HSTR     136            // padded f16 row stride (272B = 68 banks -> conflict-free 16-row reads)
#define XBUFSZ   (16 * HSTR)    // one x staging buffer (16 rows)

typedef __attribute__((ext_vector_type(16))) _Float16 v16h;
typedef __attribute__((ext_vector_type(8)))  float    v8f;

__device__ __forceinline__ v8f wmma16(v16h a, v16h b, v8f c) {
  // D(16x16,f32) = A(16x32,f16) * B(32x16,f16) + C
  return __builtin_amdgcn_wmma_f32_16x16x32_f16(false, a, false, b, (short)0, c, false, false);
}

// Copy one 64-float slice of an x row into LDS as f16 (packed 16B stores).
__device__ __forceinline__ void stage_x(const float* __restrict__ xr, _Float16* __restrict__ dst) {
#pragma unroll
  for (int e = 0; e < 64; e += 8) {
    const float4 v0 = *(const float4*)(xr + e);
    const float4 v1 = *(const float4*)(xr + e + 4);
    union { _Float16 h[8]; uint4 u; } pk;
    pk.h[0] = (_Float16)v0.x; pk.h[1] = (_Float16)v0.y;
    pk.h[2] = (_Float16)v0.z; pk.h[3] = (_Float16)v0.w;
    pk.h[4] = (_Float16)v1.x; pk.h[5] = (_Float16)v1.y;
    pk.h[6] = (_Float16)v1.z; pk.h[7] = (_Float16)v1.w;
    *(uint4*)(dst + e) = pk.u;
  }
}

__global__ void __launch_bounds__(256) zero_kernel(float* __restrict__ p, int n) {
  int i = blockIdx.x * blockDim.x + threadIdx.x;
  if (i < n) p[i] = 0.0f;
}

// Grid: 16 blocks x 256 threads. blockIdx>>3 = direction, blockIdx&7 = 16-row batch tile.
// Persistent: each block runs the whole T=1024 recurrence for its tile.
// Wave roles per step: wave0 = WMMA GEMMs, wave1 = stage x[t+1] -> LDS f16,
// wave2 = prefetch x[t+2] into cache. All waves run the VALU gate/VQC phases.
__global__ void __launch_bounds__(256) qbigru_kernel(
    const float* __restrict__ x,     const float* __restrict__ W_in,
    const float* __restrict__ b_in,  const float* __restrict__ W_out,
    const float* __restrict__ b_out,
    const float* __restrict__ wr_f,  const float* __restrict__ wr1_f,
    const float* __restrict__ wu_f,
    const float* __restrict__ wr_b,  const float* __restrict__ wr1_b,
    const float* __restrict__ wu_b,
    float* __restrict__ out)
{
  __shared__ __align__(16) _Float16 hA[16 * HSTR];     // h (then r*h) tile, f16, A-operand source
  __shared__ __align__(16) _Float16 xbuf[2 * XBUFSZ];  // double-buffered x_t tile, f16
  __shared__ float    y_s[16 * QDIM];                  // y / v1 tile (C-fragment spill for VQC)
  __shared__ float    Mre[3][QDIM][QDIM + 1];          // VQC matrices Re (padded stride 17)
  __shared__ float    Mim[3][QDIM][QDIM + 1];          // VQC matrices Im
  __shared__ float    Wout_s[4 * HIDDEN];
  __shared__ float    bout_s[HIDDEN];

  const int tid      = threadIdx.x;
  const int dir      = blockIdx.x >> 3;
  const int tileBase = (blockIdx.x & 7) * 16;

  // ---- stage constants into LDS ----
  for (int i = tid; i < 4 * HIDDEN; i += 256) Wout_s[i] = W_out[i];
  for (int i = tid; i < HIDDEN;     i += 256) bout_s[i] = b_out[i];

  // ---- build the 3 combined VQC matrices M = RX_kron @ P_cnot for this direction ----
  const float* wset0 = dir ? wr_b  : wr_f;    // reset  -> g_r
  const float* wset1 = dir ? wr1_b : wr1_f;   // reset1 -> g_r1
  const float* wset2 = dir ? wu_b  : wu_f;    // update -> g_u
  for (int idx = tid; idx < 3 * 256; idx += 256) {
    const int set = idx >> 8, rem = idx & 255;
    const int a = rem >> 4, s = rem & 15;
    const float* w = (set == 0) ? wset0 : (set == 1) ? wset1 : wset2;
    float cw[4], sw[4];
#pragma unroll
    for (int i = 0; i < 4; ++i) { cw[i] = cosf(0.5f * w[i]); sw[i] = sinf(0.5f * w[i]); }
    // CNOT ladder permutation pi(s): for i in 1..2, j in 0..3: CNOT(j -> (j+i)%4), wire0 = MSB
    int ps = s;
#pragma unroll
    for (int i = 1; i <= 2; ++i)
#pragma unroll
      for (int j = 0; j < 4; ++j) {
        const int cb = 3 - j, tb = 3 - ((j + i) & 3);
        if ((ps >> cb) & 1) ps ^= (1 << tb);
      }
    const int d = __popc(a ^ ps);
    float mag = 1.0f;
#pragma unroll
    for (int i = 0; i < 4; ++i) {
      const int bit = 3 - i;
      mag *= (((a >> bit) & 1) == ((ps >> bit) & 1)) ? cw[i] : sw[i];
    }
    float re = 0.f, im = 0.f;                 // phase (-i)^d
    switch (d & 3) { case 0: re = mag; break; case 1: im = -mag; break;
                     case 2: re = -mag; break; default: im = mag; break; }
    Mre[set][a][s] = re;
    Mim[set][a][s] = im;
  }

  // ---- per-thread ownership: batch row b_local (0..15), 8 hidden slots ----
  const int b_local = tid >> 4;
  const int aidx    = tid & 15;
  const int jb      = aidx * 8;
  float h_reg[8];
#pragma unroll
  for (int k = 0; k < 8; ++k) h_reg[k] = 0.f;
#pragma unroll
  for (int k = 0; k < 8; ++k) hA[b_local * HSTR + jb + k] = (_Float16)0.f;

  // ---- lane geometry (used by waves 0/1/2) ----
  const int wv   = tid >> 5;
  const int l    = tid & 31;
  const int m    = l & 15;       // A row / C column / B column within halves
  const int half = l >> 4;
  const int kbA  = half * 8;     // A-fragment K base per half-wave
  const int kbB  = half * 16;    // unused; B fragments also load with kbA

  // ---- wave0 WMMA state: W_in B-fragments pinned in VGPRs ----
  v16h  Bfrag[8];                // 0..3: W_in rows 0..127 (h part); 4..7: rows 128..255 (x part)
  float biasC = 0.f;
  if (wv == 0) {
#pragma unroll
    for (int c = 0; c < 8; ++c)
#pragma unroll
      for (int e = 0; e < 16; ++e)
        Bfrag[c][e] = (_Float16)W_in[(32 * c + kbA + (e & 7) + ((e >> 3) << 4)) * QDIM + m];
    biasC = b_in[m];
  } else if (wv == 1) {
    // prime x buffer 0 with the first timestep
    const int t0 = dir ? (SEQ - 1) : 0;
    stage_x(x + ((size_t)(tileBase + m) * SEQ + t0) * INPUT_SZ + half * 64,
            xbuf + m * HSTR + half * 64);
  }
  __syncthreads();

  v8f accx = {};   // bias + x_t @ W_in_x, reused by both per-step GEMMs (wave0 only)

  for (int t = 0; t < SEQ; ++t) {
    // ============ phase A: wave0 GEMM1, wave1 stage x[t+1], wave2 prefetch x[t+2] ============
    if (wv == 0) {
      const _Float16* xrow = xbuf + (t & 1) * XBUFSZ + m * HSTR;
      const _Float16* hrow = hA + m * HSTR;
      // x-part + bias (LDS, staged last step)
#pragma unroll
      for (int v = 0; v < 8; ++v) accx[v] = biasC;
#pragma unroll
      for (int c = 0; c < 4; ++c) {
        const int k0 = 32 * c + kbA;
        v16h a;
#pragma unroll
        for (int e = 0; e < 8; ++e) { a[e] = xrow[k0+e]; a[8+e] = xrow[k0+16+e]; }
        accx = wmma16(a, Bfrag[4 + c], accx);
      }
      // h-part
      v8f acch = {};
#pragma unroll
      for (int c = 0; c < 4; ++c) {
        const int k0 = 32 * c + kbA;
        v16h a;
#pragma unroll
        for (int e = 0; e < 8; ++e) { a[e] = hrow[k0+e]; a[8+e] = hrow[k0+16+e]; }
        acch = wmma16(a, Bfrag[c], acch);
      }
#pragma unroll
      for (int v = 0; v < 8; ++v)                    // spill y tile (C layout -> canonical)
        y_s[(v + 8 * half) * QDIM + m] = accx[v] + acch[v];
    } else if (wv == 1 && t + 1 < SEQ) {
      const int txn = dir ? (SEQ - 2 - t) : (t + 1);
      stage_x(x + ((size_t)(tileBase + m) * SEQ + txn) * INPUT_SZ + half * 64,
              xbuf + ((t + 1) & 1) * XBUFSZ + m * HSTR + half * 64);
    } else if (wv == 2 && t + 2 < SEQ) {
      const int txn2 = dir ? (SEQ - 3 - t) : (t + 2);
      const float* pr = x + ((size_t)(tileBase + m) * SEQ + txn2) * INPUT_SZ + half * 64;
      __builtin_prefetch(pr, 0, 0);
      __builtin_prefetch(pr + 32, 0, 0);
    }
    __syncthreads();

    // ============ phase B+C (all): VQC(y) -> g_r,g_u in regs; r; hA <- (r*h) f16 ============
    float guv[4];                                    // g_update, carried to phase E+F
    {
      float yv[16];
#pragma unroll
      for (int s = 0; s < 16; ++s) yv[s] = y_s[b_local * QDIM + s];
      float ss = 0.f;
#pragma unroll
      for (int s = 0; s < 16; ++s) ss += yv[s] * yv[s];
      const float inv = 1.0f / ss;                   // |psi|^2 normalization folded in
      float grv[4];
#pragma unroll
      for (int pass = 0; pass < 2; ++pass) {
        const int ms = (pass == 0) ? 0 : 2;
        float re = 0.f, im = 0.f;
#pragma unroll
        for (int s = 0; s < 16; ++s) {
          re += yv[s] * Mre[ms][aidx][s];
          im += yv[s] * Mim[ms][aidx][s];
        }
        const float p = (re * re + im * im) * inv;
        float g0 = (aidx & 8) ? -p : p;              // PauliZ signs, wire0 = MSB
        float g1 = (aidx & 4) ? -p : p;
        float g2 = (aidx & 2) ? -p : p;
        float g3 = (aidx & 1) ? -p : p;
#pragma unroll
        for (int msk = 1; msk <= 8; msk <<= 1) {     // butterfly: every lane gets the full sum
          g0 += __shfl_xor(g0, msk);
          g1 += __shfl_xor(g1, msk);
          g2 += __shfl_xor(g2, msk);
          g3 += __shfl_xor(g3, msk);
        }
        float* gp = (pass == 0) ? grv : guv;
        gp[0] = g0; gp[1] = g1; gp[2] = g2; gp[3] = g3;
      }
      // r = sigmoid(g_r @ W_out + b_out); write r*h back as next A operand
#pragma unroll
      for (int k = 0; k < 8; ++k) {
        const int j = jb + k;
        float acc = bout_s[j];
#pragma unroll
        for (int w = 0; w < 4; ++w) acc += grv[w] * Wout_s[w * HIDDEN + j];
        const float r = 1.0f / (1.0f + __expf(-acc));
        hA[b_local * HSTR + j] = (_Float16)(r * h_reg[k]);
      }
    }
    __syncthreads();

    // ============ phase D (wave0): v1 = (r*h) @ W_in_h + (bias + x part) ============
    if (wv == 0) {
      v8f accr = {};
      const _Float16* hrow = hA + m * HSTR;
#pragma unroll
      for (int c = 0; c < 4; ++c) {
        const int k0 = 32 * c + kbA;
        v16h a;
#pragma unroll
        for (int e = 0; e < 8; ++e) { a[e] = hrow[k0+e]; a[8+e] = hrow[k0+16+e]; }
        accr = wmma16(a, Bfrag[c], accr);
      }
#pragma unroll
      for (int v = 0; v < 8; ++v)
        y_s[(v + 8 * half) * QDIM + m] = accx[v] + accr[v];
    }
    __syncthreads();

    // ============ phase E+F (all): VQC(v1) -> g_r1; h2, z, GRU blend, write-out ============
    {
      float yv[16];
#pragma unroll
      for (int s = 0; s < 16; ++s) yv[s] = y_s[b_local * QDIM + s];
      float ss = 0.f;
#pragma unroll
      for (int s = 0; s < 16; ++s) ss += yv[s] * yv[s];
      const float inv = 1.0f / ss;
      float re = 0.f, im = 0.f;
#pragma unroll
      for (int s = 0; s < 16; ++s) {
        re += yv[s] * Mre[1][aidx][s];
        im += yv[s] * Mim[1][aidx][s];
      }
      const float p = (re * re + im * im) * inv;
      float g0 = (aidx & 8) ? -p : p;
      float g1 = (aidx & 4) ? -p : p;
      float g2 = (aidx & 2) ? -p : p;
      float g3 = (aidx & 1) ? -p : p;
#pragma unroll
      for (int msk = 1; msk <= 8; msk <<= 1) {
        g0 += __shfl_xor(g0, msk);
        g1 += __shfl_xor(g1, msk);
        g2 += __shfl_xor(g2, msk);
        g3 += __shfl_xor(g3, msk);
      }
      const float g1v[4] = { g0, g1, g2, g3 };       // g_r1, all lanes have it
#pragma unroll
      for (int k = 0; k < 8; ++k) {
        const int j = jb + k;
        float a1 = bout_s[j], a2 = bout_s[j];
#pragma unroll
        for (int w = 0; w < 4; ++w) {
          const float wvv = Wout_s[w * HIDDEN + j];
          a1 += g1v[w] * wvv;
          a2 += guv[w] * wvv;
        }
        const float h2 = tanhf(a1);
        const float z  = 1.0f / (1.0f + __expf(-a2));
        const float hn = (1.0f - z) * h_reg[k] + z * h2;
        h_reg[k] = hn;
        hA[b_local * HSTR + j] = (_Float16)hn;
        // fwd + bwd both land in slot t (reference adds without re-flipping); 0.5x each.
        unsafeAtomicAdd(out + ((size_t)(tileBase + b_local) * SEQ + t) * HIDDEN + j, 0.5f * hn);
      }
    }
    __syncthreads();
  }
}

extern "C" void kernel_launch(void* const* d_in, const int* in_sizes, int n_in,
                              void* d_out, int out_size, void* d_ws, size_t ws_size,
                              hipStream_t stream) {
  (void)in_sizes; (void)n_in; (void)d_ws; (void)ws_size;
  const float* x     = (const float*)d_in[0];
  const float* W_in  = (const float*)d_in[1];
  const float* b_in  = (const float*)d_in[2];
  const float* W_out = (const float*)d_in[3];
  const float* b_out = (const float*)d_in[4];
  const float* wr_f  = (const float*)d_in[5];
  const float* wr1_f = (const float*)d_in[6];
  const float* wu_f  = (const float*)d_in[7];
  const float* wr_b  = (const float*)d_in[8];
  const float* wr1_b = (const float*)d_in[9];
  const float* wu_b  = (const float*)d_in[10];
  float* out = (float*)d_out;

  zero_kernel<<<(out_size + 255) / 256, 256, 0, stream>>>(out, out_size);
  qbigru_kernel<<<16, 256, 0, stream>>>(x, W_in, b_in, W_out, b_out,
                                        wr_f, wr1_f, wu_f, wr_b, wr1_b, wu_b, out);
}
